// TokenAttentionLayer_66348654789151
// MI455X (gfx1250) — matrix operand order repaired and checked by reading, not compiled, not run on hardware
//
#include <hip/hip_runtime.h>

// GAT layer: B=16, L=256, Fin=128, Fout=64, N=4096
// Phase 1: Wh = h@W, f1/f2 score projections (pre-scaled by log2 e),
//          WhlT = (Wh*level)^T in bf16.
// Phase 1b: global max of f2 (for a fixed per-row softmax bound).
// Phase 2: fused masked softmax + P@Whl via v_wmma_f32_16x16x32_bf16,
//          with a FIXED per-row max (separable scores) -> no online rescale,
//          no in-loop cross-lane traffic.

#define ALPHA   0.2f
#define N_TOK   4096
#define FIN     128
#define FOUT    64
#define LOG2E   1.4426950408889634f

typedef __attribute__((ext_vector_type(16))) __bf16 v16bf;
typedef __attribute__((ext_vector_type(4)))  __bf16 v4bf;
typedef __attribute__((ext_vector_type(8)))  float  v8f;

#if __has_builtin(__builtin_amdgcn_exp2f)
#define EXP2F(x) __builtin_amdgcn_exp2f(x)
#else
#define EXP2F(x) exp2f(x)
#endif

// ---------------------------------------------------------------------------
// Kernel 1: per block = 4 rows, 256 threads (thread = (row_local, fout)).
// ---------------------------------------------------------------------------
__global__ __launch_bounds__(256) void gat_prep_kernel(
    const float* __restrict__ h, const float* __restrict__ level,
    const float* __restrict__ W, const float* __restrict__ a,
    __bf16* __restrict__ whlT, float* __restrict__ f1, float* __restrict__ f2) {
  __shared__ float ldsW[FIN * FOUT];
  __shared__ float ldsA[2 * FOUT];
  __shared__ float red1[256];
  __shared__ float red2[256];
  __shared__ float stg[4][FOUT];

  const int tid = threadIdx.x;
  for (int i = tid; i < FIN * FOUT; i += 256) ldsW[i] = W[i];
  if (tid < 2 * FOUT) ldsA[tid] = a[tid];
  __syncthreads();

  const int rloc = tid >> 6;     // 0..3
  const int f    = tid & 63;     // 0..63
  const int grow = blockIdx.x * 4 + rloc;

  const float* hrow = h + (size_t)grow * FIN;
  float acc = 0.f;
#pragma unroll 8
  for (int k = 0; k < FIN; ++k) acc += hrow[k] * ldsW[k * FOUT + f];

  const float whl = acc * level[grow];
  stg[rloc][f] = whl;
  red1[tid] = acc * ldsA[f];
  red2[tid] = acc * ldsA[FOUT + f];

  for (int s = 32; s > 0; s >>= 1) {
    __syncthreads();
    if (f < s) { red1[tid] += red1[tid + s]; red2[tid] += red2[tid + s]; }
  }
  __syncthreads();
  // Pre-scale by log2(e): leakyrelu commutes with positive scaling, so the
  // attention kernel can use raw v_exp_f32 (exp2) with no extra multiply.
  if (f == 0) { f1[grow] = red1[tid] * LOG2E; f2[grow] = red2[tid] * LOG2E; }

  // Transposed bf16 store: one 8-byte store of 4 consecutive rows per fout.
  if (rloc == 0) {
    v4bf pk;
    pk[0] = (__bf16)stg[0][f];
    pk[1] = (__bf16)stg[1][f];
    pk[2] = (__bf16)stg[2][f];
    pk[3] = (__bf16)stg[3][f];
    *(v4bf*)(whlT + (size_t)f * N_TOK + (size_t)blockIdx.x * 4) = pk;
  }
}

// ---------------------------------------------------------------------------
// Kernel 1b: global max of f2 (deterministic tree reduction, 1 block).
// ---------------------------------------------------------------------------
__global__ __launch_bounds__(256) void f2max_kernel(
    const float* __restrict__ f2, float* __restrict__ f2max) {
  __shared__ float red[256];
  const int tid = threadIdx.x;
  float m = -INFINITY;
  for (int i = tid; i < N_TOK; i += 256) m = fmaxf(m, f2[i]);
  red[tid] = m;
  for (int s = 128; s > 0; s >>= 1) {
    __syncthreads();
    if (tid < s) red[tid] = fmaxf(red[tid], red[tid + s]);
  }
  __syncthreads();
  if (tid == 0) f2max[0] = red[0];
}

// ---------------------------------------------------------------------------
// Kernel 2: fused masked softmax + P@Whl with fixed per-row max.
// Block = 128 threads = 4 wave32s; block owns 16 query rows; each wave owns
// a 1024-column K-range (split-K), combined by plain summation through LDS.
// ---------------------------------------------------------------------------
__global__ __launch_bounds__(128) void gat_attn_kernel(
    const float* __restrict__ f1, const float* __restrict__ f2,
    const float* __restrict__ f2max, const int* __restrict__ adj,
    const __bf16* __restrict__ whlT, float* __restrict__ out) {
  __shared__ float lds_l[64], lds_gl[16];
  __shared__ float lds_part[4][16 * FOUT];

  const int tid     = threadIdx.x;
  const int wave    = tid >> 5;
  const int lane    = tid & 31;
  const int laneHi  = lane >> 4;              // which K-half this lane holds
  const int row16   = lane & 15;              // A-frag row (and B-frag column)
  const int r0      = blockIdx.x * 16;
  const int grow    = r0 + row16;
  const int kbA     = laneHi ? 16 : 0;        // A frag: K 0-15 vs 16-31
  const int kbB     = laneHi ? 16 : 0;        // B frag: K 0-15 vs 16-31
  const int baseRow = laneHi ? 8  : 0;        // accumulator rows this lane holds

  const float f1r = f1[grow];
  // Fixed upper bound of every unmasked score in this row (monotone leakyrelu):
  const float fm   = f1r + f2max[0];
  const float Mrow = fmaxf(fm, ALPHA * fm);

  const int* adjRow = adj + (size_t)grow * N_TOK;

  v8f acc0 = {}, acc1 = {}, acc2 = {}, acc3 = {};
  float lrun = 0.f;

  const int jbase = wave * (N_TOK / 4);
  const int ITERS = (N_TOK / 4) / 32;

  for (int it = 0; it < ITERS; ++it) {
    const int j0 = jbase + it * 32;
    const int kb = j0 + kbA;
    if (it + 1 < ITERS) __builtin_prefetch(adjRow + kb + 32, 0, 0);

    // Probabilities directly in the WMMA 16-bit A-fragment element order.
    float p[16];
#pragma unroll
    for (int e = 0; e < 8; ++e) {
      const float raw = f1r + f2[kb + e];
      const float lr  = fmaxf(raw, ALPHA * raw);      // leakyrelu (log2 domain)
      const float ex  = EXP2F(lr - Mrow);
      p[e] = (adjRow[kb + e] > 0) ? ex : 0.f;
    }
#pragma unroll
    for (int e = 8; e < 16; ++e) {                    // memory index kb+8..kb+15
      const float raw = f1r + f2[kb + e];
      const float lr  = fmaxf(raw, ALPHA * raw);
      const float ex  = EXP2F(lr - Mrow);
      p[e] = (adjRow[kb + e] > 0) ? ex : 0.f;
    }

    v16bf aP;
#pragma unroll
    for (int e = 0; e < 16; ++e) { lrun += p[e]; aP[e] = (__bf16)p[e]; }

    // B fragments from transposed bf16 Whl: lane = column, contiguous K.
    const __bf16* bbase = whlT + j0 + kbB;
    const v16bf b0 = *(const v16bf*)(bbase + (size_t)(0 * 16 + row16) * N_TOK);
    const v16bf b1 = *(const v16bf*)(bbase + (size_t)(1 * 16 + row16) * N_TOK);
    const v16bf b2 = *(const v16bf*)(bbase + (size_t)(2 * 16 + row16) * N_TOK);
    const v16bf b3 = *(const v16bf*)(bbase + (size_t)(3 * 16 + row16) * N_TOK);

    acc0 = __builtin_amdgcn_wmma_f32_16x16x32_bf16(false, aP, false, b0, (short)0, acc0, false, false);
    acc1 = __builtin_amdgcn_wmma_f32_16x16x32_bf16(false, aP, false, b1, (short)0, acc1, false, false);
    acc2 = __builtin_amdgcn_wmma_f32_16x16x32_bf16(false, aP, false, b2, (short)0, acc2, false, false);
    acc3 = __builtin_amdgcn_wmma_f32_16x16x32_bf16(false, aP, false, b3, (short)0, acc3, false, false);
  }

  // ---- cross-wave split-K combine: plain sums (shared fixed max) ----
  lrun += __shfl_xor(lrun, 16, 32);            // combine K-half lane pair
  if (lane < 16) lds_l[wave * 16 + lane] = lrun;
  __syncthreads();
  if (tid < 16) {
    const float gl = lds_l[tid] + lds_l[16 + tid] + lds_l[32 + tid] + lds_l[48 + tid];
    lds_gl[tid] = fmaxf(gl, 1e-37f);           // guard (all-masked row ~ impossible)
  }
  __syncthreads();

#pragma unroll
  for (int v = 0; v < 8; ++v) {
    const int M = baseRow + v;
    lds_part[wave][M * 64 + 0 * 16 + row16] = acc0[v];
    lds_part[wave][M * 64 + 1 * 16 + row16] = acc1[v];
    lds_part[wave][M * 64 + 2 * 16 + row16] = acc2[v];
    lds_part[wave][M * 64 + 3 * 16 + row16] = acc3[v];
  }
  __syncthreads();

#pragma unroll
  for (int i = 0; i < 8; ++i) {
    const int o   = i * 128 + tid;
    const int row = o >> 6;
    float val = lds_part[0][o] + lds_part[1][o] + lds_part[2][o] + lds_part[3][o];
    val /= lds_gl[row];
    val = val > 0.f ? val : ALPHA * val;
    out[(size_t)(r0 + row) * FOUT + (o & 63)] = val;
  }
}

// ---------------------------------------------------------------------------
extern "C" void kernel_launch(void* const* d_in, const int* in_sizes, int n_in,
                              void* d_out, int out_size, void* d_ws, size_t ws_size,
                              hipStream_t stream) {
  const float* h     = (const float*)d_in[0];
  const int*   adj   = (const int*)d_in[1];
  const float* level = (const float*)d_in[2];
  const float* W     = (const float*)d_in[3];
  const float* a     = (const float*)d_in[4];
  float* out = (float*)d_out;

  // Workspace: WhlT bf16 [64][4096] (512KB), f1[4096], f2[4096], f2max[1].
  __bf16* whlT  = (__bf16*)d_ws;
  float*  f1    = (float*)((char*)d_ws + (size_t)FOUT * N_TOK * sizeof(__bf16));
  float*  f2    = f1 + N_TOK;
  float*  f2max = f2 + N_TOK;

  gat_prep_kernel<<<N_TOK / 4, 256, 0, stream>>>(h, level, W, a, whlT, f1, f2);
  f2max_kernel<<<1, 256, 0, stream>>>(f2, f2max);
  gat_attn_kernel<<<N_TOK / 16, 128, 0, stream>>>(f1, f2, f2max, adj, whlT, out);
}
